// MarginalFitnessModel_33767032881397
// MI455X (gfx1250) — matrix operand order repaired and checked by reading, not compiled, not run on hardware
//
#include <hip/hip_runtime.h>
#include <cstddef>

#define N_GT    20000
#define KPAD    20096          // 157 * 128, zero-padded K for the bf16 A copy
#define N_T     50
#define N_OUT   8000           // N_POS * ALPHA
#define N_POSF  2000.0f
#define MPAD    64             // N_T padded to 4 WMMA row-tiles

#define KC       128           // K elements staged per pipeline stage
#define NSTAGES  (KPAD / KC)   // 157
#define KSPLIT   4             // K split across blockIdx.y for occupancy
#define NTILE    64            // N columns per block
#define SROW     136           // LDS row stride in bf16 (128 + 8 pad): 272 B, 16B-aligned
#define PSTRIDE  ((size_t)N_T * N_OUT)   // partial buffer stride (400000)

typedef __attribute__((ext_vector_type(16))) __bf16 v16bf;
typedef __attribute__((ext_vector_type(8)))  float  v8f;

// packed f32x2 -> bf16x2 (RNE) in one VALU op
static __device__ __forceinline__ unsigned int pack_bf16(float lo, float hi) {
    unsigned int r;
    asm("v_cvt_pk_bf16_f32 %0, %1, %2" : "=v"(r) : "v"(lo), "v"(hi));
    return r;
}

// ---------------------------------------------------------------------------
// Kernel 1: row softmax -> fqm (f32) + bf16 copy zero-padded to [64][20096],
//           plus per-row unnormalized skew.
// ---------------------------------------------------------------------------
__global__ void softmax_skew_kernel(const float* __restrict__ fq,
                                    float* __restrict__ fqm,
                                    unsigned short* __restrict__ fqmh,
                                    float* __restrict__ skew) {
    const int t   = blockIdx.x;
    const int tid = threadIdx.x;
    __shared__ float sdata[256];

    unsigned short* hrow = fqmh + (size_t)t * KPAD;
    if (t >= N_T) {                       // zero rows 50..63 entirely
        for (int j = tid; j < KPAD; j += 256) hrow[j] = 0;
        return;
    }
    for (int j = N_GT + tid; j < KPAD; j += 256) hrow[j] = 0;  // K padding tail

    const float* x = fq + (size_t)t * N_GT;

    // pass 1: max
    float mx = -3.0e38f;
    for (int j = tid; j < N_GT; j += 256) mx = fmaxf(mx, x[j]);
    sdata[tid] = mx; __syncthreads();
    for (int s = 128; s > 0; s >>= 1) {
        if (tid < s) sdata[tid] = fmaxf(sdata[tid], sdata[tid + s]);
        __syncthreads();
    }
    mx = sdata[0]; __syncthreads();

    // pass 2: sum of exp
    float se = 0.0f;
    for (int j = tid; j < N_GT; j += 256) se += expf(x[j] - mx);
    sdata[tid] = se; __syncthreads();
    for (int s = 128; s > 0; s >>= 1) {
        if (tid < s) sdata[tid] += sdata[tid + s];
        __syncthreads();
    }
    const float inv = 1.0f / sdata[0]; __syncthreads();

    // pass 3: write fqm (f32 + packed bf16 pairs), accumulate skew
    float* frow = fqm + (size_t)t * N_GT;
    const float m = 1.0f / (float)N_GT;
    float sk = 0.0f;
    for (int j = tid * 2; j < N_GT; j += 512) {
        float f0 = expf(x[j]     - mx) * inv;
        float f1 = expf(x[j + 1] - mx) * inv;
        frow[j]     = f0;
        frow[j + 1] = f1;
        *(unsigned int*)(hrow + j) = pack_bf16(f0, f1);
        float d0 = f0 - m, d1 = f1 - m;
        sk += d0 * d0 * d0 + d1 * d1 * d1;
    }
    sdata[tid] = sk; __syncthreads();
    for (int s = 128; s > 0; s >>= 1) {
        if (tid < s) sdata[tid] += sdata[tid + s];
        __syncthreads();
    }
    if (tid == 0) skew[t] = sdata[0] * (1.0f / (float)N_GT);
}

// ---------------------------------------------------------------------------
// GEMM staging helpers: 64N x 128K f32 tile per stage, 128 threads.
// Thread: nn4 = (tid&15)*4 (4 consecutive N), kq = tid>>4 (16 consecutive K).
// GUARD path only needed for the very last K stage (K 19968..20096).
// ---------------------------------------------------------------------------
union F4 { float4 v; float f[4]; };

template <bool GUARD>
static __device__ __forceinline__ void stage_load(const float* __restrict__ geno,
                                                  int k0, int n0, int nn4, int kq,
                                                  F4 g[16]) {
    const int kbase = k0 + kq * 16;
    const float4* gp = (const float4*)(geno + (size_t)kbase * N_OUT + n0 + nn4);
#pragma unroll
    for (int r = 0; r < 16; ++r) {
        if (!GUARD || (kbase + r < N_GT)) g[r].v = gp[(size_t)r * (N_OUT / 4)];
        else                              g[r].v = make_float4(0.f, 0.f, 0.f, 0.f);
    }
}

static __device__ __forceinline__ void stage_store(unsigned short* __restrict__ Bsb,
                                                   int nn4, int kq, const F4 g[16]) {
#pragma unroll
    for (int c = 0; c < 4; ++c) {
        uint4 lo, hi;
        lo.x = pack_bf16(g[0].f[c],  g[1].f[c]);
        lo.y = pack_bf16(g[2].f[c],  g[3].f[c]);
        lo.z = pack_bf16(g[4].f[c],  g[5].f[c]);
        lo.w = pack_bf16(g[6].f[c],  g[7].f[c]);
        hi.x = pack_bf16(g[8].f[c],  g[9].f[c]);
        hi.y = pack_bf16(g[10].f[c], g[11].f[c]);
        hi.z = pack_bf16(g[12].f[c], g[13].f[c]);
        hi.w = pack_bf16(g[14].f[c], g[15].f[c]);
        uint4* dst = (uint4*)(Bsb + (size_t)(nn4 + c) * SROW + kq * 16);
        dst[0] = lo;
        dst[1] = hi;
    }
}

// ---------------------------------------------------------------------------
// Kernel 2: partial GEMM  part[z] = fqm_bf16[64 x KPAD(z-slice)] @ geno
// 128 threads = 4 waves; wave = M-tile; 4 N-subtile accumulators per wave
// (A fragment reused 4x). Double-buffered LDS, 16 WMMAs per stage, 1 barrier.
// blockIdx.x: N tile (64 cols), blockIdx.y: K split segment.
// ---------------------------------------------------------------------------
__global__ void gemm_partial_kernel(const unsigned short* __restrict__ fqmh,
                                    const float* __restrict__ geno,
                                    float* __restrict__ partials) {
    const int tid  = threadIdx.x;
    const int lane = tid & 31;
    const int mt   = tid >> 5;            // wave id == M tile
    const int n0   = blockIdx.x * NTILE;
    const int z    = blockIdx.y;

    const int sBeg = (NSTAGES * z) / KSPLIT;
    const int sEnd = (NSTAGES * (z + 1)) / KSPLIT;

    __shared__ unsigned short Bs[2][NTILE * SROW];   // 2 x 17408 B

    const int hiHalf = (lane >= 16) ? 1 : 0;
    const int mrow   = mt * 16 + (lane & 15);

    // staging coordinates
    const int nn4 = (tid & 15) * 4;
    const int kq  = tid >> 4;             // 0..7 -> k rows [16*kq, 16*kq+16)

    // fragment bases
    const int kb = hiHalf ? 8 : 0;        // A K-base within a 32-K step
    const uint4* aBase = (const uint4*)(fqmh + (size_t)mrow * KPAD + kb);
    const size_t bOff  = (size_t)(lane & 15) * SROW + (hiHalf ? 8 : 0); // elems

    v8f acc[4];
#pragma unroll
    for (int j = 0; j < 4; ++j) acc[j] = (v8f){0.f,0.f,0.f,0.f,0.f,0.f,0.f,0.f};

    // prologue: stage first tile of this segment (never the final K stage)
    {
        F4 g[16];
        stage_load<false>(geno, sBeg * KC, n0, nn4, kq, g);
        stage_store(&Bs[0][0], nn4, kq, g);
    }
    __syncthreads();

    for (int s = sBeg; s < sEnd; ++s) {
        const int k0  = s * KC;
        const int buf = (s - sBeg) & 1;
        const bool havenext = (s + 1 < sEnd);

        // (a) next tile's global loads issued early (overlap with WMMAs);
        //     guarded path only when staging the final K stage
        F4 g[16];
        if (havenext) {
            if (s + 2 == NSTAGES) stage_load<true >(geno, k0 + KC, n0, nn4, kq, g);
            else                  stage_load<false>(geno, k0 + KC, n0, nn4, kq, g);
        }

        // (b) 4 K-steps x 4 N-subtiles from current LDS buffer
        const uint4* ap = aBase + (size_t)(k0 >> 3);          // k0*2 bytes / 16
        const uint4* bp = (const uint4*)(&Bs[buf][0] + bOff);
#pragma unroll
        for (int q = 0; q < 4; ++q) {
            union { uint4 u[2]; v16bf v; } A_;
            A_.u[0] = ap[q * 4 + 0];          // K = kb      .. kb+7
            A_.u[1] = ap[q * 4 + 2];          // K = kb + 16 .. kb+23
            // preload all 4 B fragments, then issue the 4 WMMAs
            union { uint4 u[2]; v16bf v; } B_[4];
#pragma unroll
            for (int j = 0; j < 4; ++j) {
                const uint4* bpj = bp + (size_t)j * (16 * SROW / 8);
                B_[j].u[0] = bpj[q * 4 + 0];  // K = kb .. kb+7, kb+16 .. kb+23 (matches A)
                B_[j].u[1] = bpj[q * 4 + 2];
            }
#pragma unroll
            for (int j = 0; j < 4; ++j)
                acc[j] = __builtin_amdgcn_wmma_f32_16x16x32_bf16(
                             false, A_.v, false, B_[j].v, (short)0, acc[j], false, false);
        }

        // (c) convert + store next tile into the other buffer
        if (havenext) stage_store(&Bs[buf ^ 1][0], nn4, kq, g);

        __syncthreads();
    }

    // epilogue: write raw partial sums for this K segment
    float* part = partials + (size_t)z * PSTRIDE;
    const int rOff = hiHalf ? 8 : 0;
#pragma unroll
    for (int j = 0; j < 4; ++j) {
        const int n = n0 + j * 16 + (lane & 15);
#pragma unroll
        for (int r = 0; r < 8; ++r) {
            int M = mt * 16 + r + rOff;
            if (M < N_T)
                part[(size_t)M * N_OUT + n] = acc[j][r];
        }
    }
}

// ---------------------------------------------------------------------------
// Kernel 2b: combine K-split partials, apply /N_POS + 1e-10 and log.
// ---------------------------------------------------------------------------
__global__ void combine_log_kernel(const float* __restrict__ partials,
                                   float* __restrict__ out) {
    const size_t i = (size_t)blockIdx.x * 256 + threadIdx.x;
    if (i < PSTRIDE) {
        float s = partials[i] + partials[PSTRIDE + i]
                + partials[2 * PSTRIDE + i] + partials[3 * PSTRIDE + i];
        out[i] = logf(s * (1.0f / N_POSF) + 1e-10f);
    }
}

// ---------------------------------------------------------------------------
// Kernel 3: per-transition weight / KL loss (closed form of the 5-step
// replicator iteration: x5 = p0m * e^{5f} / S,  S = sum(p0m * e^{5f})).
// ---------------------------------------------------------------------------
__global__ void transition_loss_kernel(const float* __restrict__ fqm,
                                       const float* __restrict__ fitness,
                                       float* __restrict__ wt,
                                       float* __restrict__ lt) {
    const int t   = blockIdx.x;          // 0..48
    const int tid = threadIdx.x;
    __shared__ float sdata[256];
    __shared__ float s_weight, s_S;

    const float* p0 = fqm + (size_t)t * N_GT;
    const float* p1 = fqm + (size_t)(t + 1) * N_GT;

    float wl = 0.0f, Sl = 0.0f;
    for (int j = tid; j < N_GT; j += 256) {
        float a = p0[j];
        if (a >= 1e-6f) {
            wl += p1[j];
            Sl += a * expf(5.0f * fitness[j]);
        }
    }
    sdata[tid] = wl; __syncthreads();
    for (int s = 128; s > 0; s >>= 1) { if (tid < s) sdata[tid] += sdata[tid + s]; __syncthreads(); }
    if (tid == 0) s_weight = sdata[0];
    __syncthreads();
    sdata[tid] = Sl; __syncthreads();
    for (int s = 128; s > 0; s >>= 1) { if (tid < s) sdata[tid] += sdata[tid + s]; __syncthreads(); }
    if (tid == 0) s_S = sdata[0];
    __syncthreads();

    const float weight = s_weight;
    const float lw = logf(weight);
    const float lS = logf(s_S);

    float kl = 0.0f;
    for (int j = tid; j < N_GT; j += 256) {
        float a = p0[j], b = p1[j];
        if (a >= 1e-6f && b > 0.0f) {
            float np1 = b / weight;
            kl += np1 * ((logf(b) - lw) - (logf(a) + 5.0f * fitness[j] - lS));
        }
    }
    sdata[tid] = kl; __syncthreads();
    for (int s = 128; s > 0; s >>= 1) { if (tid < s) sdata[tid] += sdata[tid + s]; __syncthreads(); }
    if (tid == 0) { wt[t] = weight; lt[t] = weight * sdata[0]; }
}

// ---------------------------------------------------------------------------
// Kernel 4: finalize the two scalar losses.
// ---------------------------------------------------------------------------
__global__ void finalize_kernel(const float* __restrict__ skew,
                                const float* __restrict__ wt,
                                const float* __restrict__ lt,
                                float* __restrict__ out) {
    if (threadIdx.x == 0) {
        float fl = 0.0f;
        for (int t = 0; t < N_T - 1; ++t)
            if (wt[t] >= 0.3f) fl += lt[t];
        out[PSTRIDE]     = fl / (float)(N_T - 1);
        float sk = 0.0f;
        for (int t = 0; t < N_T; ++t) sk += skew[t];
        out[PSTRIDE + 1] = -sk / (float)N_T;
    }
}

// ---------------------------------------------------------------------------
extern "C" void kernel_launch(void* const* d_in, const int* in_sizes, int n_in,
                              void* d_out, int out_size, void* d_ws, size_t ws_size,
                              hipStream_t stream) {
    const float* fitness = (const float*)d_in[0];   // (20000,)
    const float* fq_mat  = (const float*)d_in[1];   // (50, 20000)
    const float* geno    = (const float*)d_in[2];   // (20000, 8000)
    float* out = (float*)d_out;                     // 400000 + 2 floats

    // workspace layout (16B-aligned blocks)
    float*          fqm   = (float*)d_ws;                                 // 50*20000 f32
    unsigned short* fqmh  = (unsigned short*)(fqm + (size_t)N_T * N_GT);  // 64*20096 bf16
    float*          parts = (float*)(fqmh + (size_t)MPAD * KPAD);         // 4*400000 f32
    float*          skew  = parts + KSPLIT * PSTRIDE;                     // 64 f32
    float*          wt    = skew + 64;                                    // 64 f32
    float*          lt    = wt + 64;                                      // 64 f32

    softmax_skew_kernel   <<<MPAD, 256, 0, stream>>>(fq_mat, fqm, fqmh, skew);
    gemm_partial_kernel   <<<dim3(N_OUT / NTILE, KSPLIT), 128, 0, stream>>>(fqmh, geno, parts);
    combine_log_kernel    <<<(int)((PSTRIDE + 255) / 256), 256, 0, stream>>>(parts, out);
    transition_loss_kernel<<<N_T - 1, 256, 0, stream>>>(fqm, fitness, wt, lt);
    finalize_kernel       <<<1, 64, 0, stream>>>(skew, wt, lt, out);
}
